// Model_89953795048036
// MI455X (gfx1250) — compile-verified, matrix operand order checked by reading, not run on hardware
//
#include <hip/hip_runtime.h>
#include <hip/hip_bf16.h>

// ---------------------------------------------------------------------------
// Persistent-kernel LSTM for MI455X (gfx1250, wave32, WMMA).
//   gates = x_t*w_in + h @ W_hh^T + bias ; LSTM cell update ; final linear.
// W_hh slice lives in LDS (bf16), c lives in registers (WMMA lane mapping is
// step-invariant), h ping-pongs through L2 in bf16, one atomic grid barrier
// per timestep. Matmul uses v_wmma_f32_16x16x32_bf16 with explicit
// double-buffered A/B fragments; activations use hardware v_tanh_f32.
// ---------------------------------------------------------------------------

typedef __attribute__((ext_vector_type(16))) __bf16    v16bf;
typedef __attribute__((ext_vector_type(8)))  float     v8f;
typedef __attribute__((ext_vector_type(4)))  unsigned  v4u;

union FragU { v16bf bf; v4u q[2]; };

#define HID        512
#define BATCH      256
#define TSTEPS     784
#define NWG        32
#define COLS       16          // hidden columns per WG
#define LDS_STRIDE 520         // 512 + 8 pad (bank-conflict avoidance), 16B aligned
#define NCLS       10

__device__ __forceinline__ unsigned short f2bf(float f) {
  unsigned u = __builtin_bit_cast(unsigned, f);
  u += 0x7FFFu + ((u >> 16) & 1u);              // round-to-nearest-even
  return (unsigned short)(u >> 16);
}
__device__ __forceinline__ float bf2f(unsigned short h) {
  unsigned u = ((unsigned)h) << 16;
  return __builtin_bit_cast(float, u);
}
// Hardware transcendental tanh (gfx1250 TRANS op). Trailing v_nop satisfies
// the 1-independent-op TRANS hazard window (ISA 7.4) before the result is read.
__device__ __forceinline__ float tanh_hw(float x) {
  float r;
  asm("v_tanh_f32 %0, %1\n\tv_nop" : "=v"(r) : "v"(x));
  return r;
}
__device__ __forceinline__ float sig_hw(float x) {
  return __builtin_fmaf(tanh_hw(0.5f * x), 0.5f, 0.5f);   // sigmoid via tanh
}

__global__ __launch_bounds__(256)
void lstm_persistent_gfx1250(const float* __restrict__ inputs,   // [256][784]
                             const float* __restrict__ W_ih,     // [2048]
                             const float* __restrict__ W_hh,     // [2048][512]
                             const float* __restrict__ b_ih,     // [2048]
                             const float* __restrict__ b_hh,     // [2048]
                             const float* __restrict__ W_lin,    // [10][512]
                             const float* __restrict__ b_lin,    // [10]
                             unsigned short* __restrict__ h0buf, // [256][512] bf16
                             unsigned short* __restrict__ h1buf, // [256][512] bf16
                             unsigned*       __restrict__ bar,
                             float*          __restrict__ out)   // [256][10]
{
  extern __shared__ char smem_raw[];
  unsigned short* Wlds = (unsigned short*)smem_raw;                    // 64 x 520 bf16
  float*          xlds = (float*)(smem_raw + 64 * LDS_STRIDE * 2);     // 256 f32

  const int tid  = threadIdx.x;
  const int wg   = blockIdx.x;
  const int lane = tid & 31;
  const int wave = tid >> 5;
  const int hi   = lane >> 4;      // half-wave select (K sub-segment)
  const int m    = lane & 15;      // A: row-in-tile / B: column-in-tile

  // ---- one-time: stage this WG's W_hh slice into LDS as bf16 --------------
  {
    int r = tid >> 2;              // 0..63 : LDS row = gate*16 + n
    int q = tid & 3;               // quarter of the 512-wide K row
    int g = r >> 4, n = r & 15;
    const float*    src = W_hh + (size_t)(g * HID + wg * COLS + n) * HID + q * 128;
    unsigned short* dst = Wlds + r * LDS_STRIDE + q * 128;
#pragma unroll 8
    for (int k = 0; k < 128; ++k) dst[k] = f2bf(src[k]);
  }

  // ---- per-lane gate constants (column = wg*16 + m, gates 0..3 = i,f,g,o) -
  float wing[4], biasg[4];
  {
    int col = wg * COLS + m;
#pragma unroll
    for (int g = 0; g < 4; ++g) {
      int gc   = g * HID + col;
      wing[g]  = W_ih[gc];
      biasg[g] = b_ih[gc] + b_hh[gc];
    }
  }

  const int mbase0 = wave * 32;        // this wave's two 16-row M tiles
  const int mbase1 = wave * 32 + 16;

  // cell state lives in registers: lane owns rows {mbase+r+8*hi}, col wg*16+m
  float cstate[2][8];
#pragma unroll
  for (int mt = 0; mt < 2; ++mt)
#pragma unroll
    for (int r = 0; r < 8; ++r) cstate[mt][r] = 0.0f;

  __syncthreads();

  unsigned short* hr = h0buf;   // read h_t   (zero-initialized by host memset)
  unsigned short* hw = h1buf;   // write h_{t+1}

  const int k0 = hi * 8;        // per-lane K sub-offset within a 32-chunk

  for (int t = 0; t < TSTEPS; ++t) {
    // stage x_t column into LDS; prefetch next column (global_prefetch_b8)
    xlds[tid] = inputs[(size_t)tid * TSTEPS + t];
    if (t + 1 < TSTEPS)
      __builtin_prefetch(&inputs[(size_t)tid * TSTEPS + t + 1], 0, 0);
    __syncthreads();

    v8f acc[4][2];
#pragma unroll
    for (int g = 0; g < 4; ++g) { acc[g][0] = (v8f){}; acc[g][1] = (v8f){}; }

    // ---- K loop: gates[256x16x4] += h[256x512] @ Whh_slice^T ------------
    // Explicit double-buffered fragments: load chunk k+1 (A from L2, B x4
    // gates from LDS) while chunk k's 8 WMMAs issue.
    const unsigned short* hrow0 = hr + (size_t)(mbase0 + m) * HID + k0;
    const unsigned short* hrow1 = hr + (size_t)(mbase1 + m) * HID + k0;
    const unsigned short* bbase = Wlds + m * LDS_STRIDE + k0;

    FragU A0[2], A1[2], B[2][4];

#define LOAD_CHUNK(buf, kc)                                                  \
    do {                                                                     \
      A0[buf].q[0] = *(const v4u*)(hrow0 + (kc));                            \
      A0[buf].q[1] = *(const v4u*)(hrow0 + (kc) + 16);                       \
      A1[buf].q[0] = *(const v4u*)(hrow1 + (kc));                            \
      A1[buf].q[1] = *(const v4u*)(hrow1 + (kc) + 16);                       \
      _Pragma("unroll")                                                      \
      for (int g = 0; g < 4; ++g) {                                          \
        const unsigned short* bp = bbase + g * 16 * LDS_STRIDE + (kc);       \
        B[buf][g].q[0] = *(const v4u*)(bp);                                  \
        B[buf][g].q[1] = *(const v4u*)(bp + 16);                             \
      }                                                                      \
    } while (0)

    LOAD_CHUNK(0, 0);
#pragma unroll
    for (int kc = 0; kc < HID; kc += 32) {
      const int cur = (kc >> 5) & 1;
      const int nxt = cur ^ 1;
      if (kc + 32 < HID) LOAD_CHUNK(nxt, kc + 32);
#pragma unroll
      for (int g = 0; g < 4; ++g) {
        acc[g][0] = __builtin_amdgcn_wmma_f32_16x16x32_bf16(
            false, A0[cur].bf, false, B[cur][g].bf, (short)0, acc[g][0], false, false);
        acc[g][1] = __builtin_amdgcn_wmma_f32_16x16x32_bf16(
            false, A1[cur].bf, false, B[cur][g].bf, (short)0, acc[g][1], false, false);
      }
    }
#undef LOAD_CHUNK

    // ---- elementwise LSTM cell update (C/D layout: M=r+8*hi, N=m) -------
#pragma unroll
    for (int mt = 0; mt < 2; ++mt) {
      const int mbase = mt ? mbase1 : mbase0;
#pragma unroll
      for (int r = 0; r < 8; ++r) {
        const int row = mbase + r + hi * 8;
        const float x  = xlds[row];
        const float gi = acc[0][mt][r] + __builtin_fmaf(x, wing[0], biasg[0]);
        const float gf = acc[1][mt][r] + __builtin_fmaf(x, wing[1], biasg[1]);
        const float gg = acc[2][mt][r] + __builtin_fmaf(x, wing[2], biasg[2]);
        const float go = acc[3][mt][r] + __builtin_fmaf(x, wing[3], biasg[3]);
        const float c  = sig_hw(gf) * cstate[mt][r] + sig_hw(gi) * tanh_hw(gg);
        cstate[mt][r]  = c;
        const float h  = sig_hw(go) * tanh_hw(c);
        hw[(size_t)row * HID + wg * COLS + m] = f2bf(h);
      }
    }

    // ---- device-wide step barrier (monotonic L2 atomic counter) ---------
    __threadfence();
    __syncthreads();
    if (tid == 0) {
      atomicAdd(bar, 1u);
      const unsigned target = (unsigned)(t + 1) * NWG;
      while (*((volatile unsigned*)bar) < target) __builtin_amdgcn_s_sleep(1);
    }
    __syncthreads();
    __threadfence();

    unsigned short* tmp = hr; hr = hw; hw = tmp;
  }

  // ---- final linear: out = h_T @ W_lin^T + b_lin (WG0, 1 row/thread) -----
  if (wg == 0) {
    const unsigned short* hrow = hr + (size_t)tid * HID;
#pragma unroll 1
    for (int cls = 0; cls < NCLS; ++cls) {
      float a = b_lin[cls];
      const float* wl = W_lin + cls * HID;
      for (int k = 0; k < HID; ++k) a += bf2f(hrow[k]) * wl[k];
      out[tid * NCLS + cls] = a;
    }
  }
}

extern "C" void kernel_launch(void* const* d_in, const int* in_sizes, int n_in,
                              void* d_out, int out_size, void* d_ws, size_t ws_size,
                              hipStream_t stream) {
  (void)in_sizes; (void)n_in; (void)out_size; (void)ws_size;
  const float* inputs = (const float*)d_in[0];
  const float* W_ih   = (const float*)d_in[1];
  const float* W_hh   = (const float*)d_in[2];
  const float* b_ih   = (const float*)d_in[3];
  const float* b_hh   = (const float*)d_in[4];
  const float* W_lin  = (const float*)d_in[5];
  const float* b_lin  = (const float*)d_in[6];

  const size_t hbytes = (size_t)BATCH * HID * sizeof(unsigned short); // 256 KB
  unsigned short* h0  = (unsigned short*)d_ws;
  unsigned short* h1  = (unsigned short*)((char*)d_ws + hbytes);
  unsigned*       bar = (unsigned*)((char*)d_ws + 2 * hbytes);

  // zero h0 (=h_0), h1 and barrier counter; graph-capture-legal async memset
  hipMemsetAsync(d_ws, 0, 2 * hbytes + 64, stream);

  const size_t smem = (size_t)64 * LDS_STRIDE * 2 + (size_t)BATCH * 4; // ~67.6 KB
  lstm_persistent_gfx1250<<<NWG, 256, smem, stream>>>(
      inputs, W_ih, W_hh, b_ih, b_hh, W_lin, b_lin, h0, h1, bar, (float*)d_out);
}
